// GraphAttentionLayer_78941498900661
// MI455X (gfx1250) — matrix operand order repaired and checked by reading, not compiled, not run on hardware
//
#include <hip/hip_runtime.h>

#define NN 4096
#define FD 64
#define BATCH 4
#define TI 64
#define TJ 64
#define PAD 33   // dword row stride in LDS tiles (32 f16-pairs + 1 pad)

typedef __attribute__((ext_vector_type(16))) _Float16 v16h;
typedef __attribute__((ext_vector_type(8)))  float    v8f;
typedef __attribute__((ext_vector_type(8)))  unsigned int v8u;
typedef __attribute__((ext_vector_type(2)))  _Float16 h2;

// ---------------------------------------------------------------------------
// Kernel 1: h = inp @ W ; lh = leakyrelu(h); f_src = lh@a[:64], f_dst = lh@a[64:]
// One 64-thread block per row (b*N + n). Also writes hT in f16 [B][64][N]
// so the WMMA B-fragment is a single packed-DWORD LDS read later.
// ---------------------------------------------------------------------------
__global__ __launch_bounds__(64) void gat_proj_kernel(
    const float* __restrict__ inp, const float* __restrict__ W,
    const float* __restrict__ a,
    _Float16* __restrict__ hT, float* __restrict__ fsrc, float* __restrict__ fdst) {
  __shared__ float row[FD];
  __shared__ float red0[FD], red1[FD];
  const int r = blockIdx.x;          // 0 .. B*N-1
  const int t = threadIdx.x;         // 0 .. 63
  const int b = r >> 12;             // / 4096
  const int n = r & (NN - 1);
  row[t] = inp[(size_t)r * FD + t];
  __syncthreads();
  float h = 0.f;
#pragma unroll
  for (int f = 0; f < FD; ++f) h = fmaf(row[f], W[f * FD + t], h);
  const float lh = h > 0.f ? h : 0.2f * h;        // leakyrelu, slope 0.2
  red0[t] = lh * a[t];
  red1[t] = lh * a[FD + t];
  hT[((size_t)(b * FD + t)) * NN + n] = (_Float16)h;
  __syncthreads();
  for (int s = 32; s > 0; s >>= 1) {
    if (t < s) { red0[t] += red0[t + s]; red1[t] += red1[t + s]; }
    __syncthreads();
  }
  if (t == 0) { fsrc[r] = red0[0]; fdst[r] = red1[0]; }
}

// ---------------------------------------------------------------------------
// Kernel 2: column-wise (axis=1) softmax stats via online max/sum.
// One thread per column j; adjacent threads read adjacent j -> coalesced.
// Single streaming pass of adj (256 MB), one fast exp per element.
// ---------------------------------------------------------------------------
__global__ __launch_bounds__(256) void gat_colstats_kernel(
    const float* __restrict__ adj, const float* __restrict__ fsrc,
    const float* __restrict__ fdst,
    float* __restrict__ colmax, float* __restrict__ colrz) {
  const int c = blockIdx.x * 256 + threadIdx.x;   // b*N + j
  const int b = c >> 12;
  const float fd = fdst[c];
  const float* acol = adj + (size_t)b * NN * NN + (c & (NN - 1));
  const float* fs = fsrc + b * NN;
  float m = -3.0e38f, Z = 0.f;
  for (int i = 0; i < NN; ++i) {
    float s = fmaf(0.8f, acol[(size_t)i * NN], 0.2f * (fs[i] + fd));
    s = s > 0.f ? s : -1.0e12f;                   // mask
    if (s > m) { Z = Z * __expf(m - s) + 1.f; m = s; }
    else       { Z += __expf(s - m); }
  }
  colmax[c] = m;
  colrz[c]  = 1.f / Z;                            // Z >= 1 always
}

// ---------------------------------------------------------------------------
// Kernel 3: h'[i,:] = relu( sum_j att[i,j] * h[j,:] ) using f16 WMMA.
// Block = 256 thr (8 waves), output tile 64 i-rows x 64 o-cols; K-loop over j
// in steps of 64 (two K=32 WMMA chunks -> 4 v_wmma per iteration).
// att tile built on the fly from adj + column stats, packed f16 -> LDS.
// ---------------------------------------------------------------------------
__global__ __launch_bounds__(256) void gat_attn_gemm_kernel(
    const float* __restrict__ adj, const unsigned int* __restrict__ hTu, // f16x2 [B][64][N/2]
    const float* __restrict__ fsrc, const float* __restrict__ fdst,
    const float* __restrict__ colmax, const float* __restrict__ colrz,
    float* __restrict__ out) {
  __shared__ unsigned int attLds[TI * PAD];  // [64 rows][32 f16-pairs] padded
  __shared__ unsigned int hLds[FD * PAD];    // [64 o-cols][32 f16-pairs] padded
  __shared__ float fsLds[TI];

  const int b  = blockIdx.y;
  const int i0 = blockIdx.x * TI;
  const int t  = threadIdx.x;
  if (t < TI) fsLds[t] = fsrc[b * NN + i0 + t];

  const int lane   = t & 31;
  const int wave   = t >> 5;
  const int hiHalf = (lane >> 4) & 1;
  const int lrow   = lane & 15;
  const int rg     = wave >> 1;          // row group 0..3 (16 rows each)
  const int cg0    = (wave & 1) * 2;     // first of two col groups

  v8f acc0 = {}; v8f acc1 = {};
  const size_t adjBase = (size_t)b * NN * NN;
  const int fbase = b * NN;

  for (int j0 = 0; j0 < NN; j0 += TJ) {
    __syncthreads();   // protect LDS from previous iteration's readers
    // ---- stage att tile [TI x TJ] as packed f16 pairs (coalesced adj reads)
#pragma unroll
    for (int p = t; p < TI * TJ / 2; p += 256) {
      const int il = p >> 5;             // row, 32 pairs per row
      const int jp = p & 31;
      const int j  = j0 + jp * 2;
      const float2 av  = *(const float2*)(adj + adjBase + (size_t)(i0 + il) * NN + j);
      const float  fsv = fsLds[il];
      const float2 fdv = *(const float2*)(fdst + fbase + j);
      const float2 mv  = *(const float2*)(colmax + fbase + j);
      const float2 rzv = *(const float2*)(colrz + fbase + j);
      float s0 = fmaf(0.8f, av.x, 0.2f * (fsv + fdv.x));
      float s1 = fmaf(0.8f, av.y, 0.2f * (fsv + fdv.y));
      s0 = s0 > 0.f ? s0 : -1.0e12f;
      s1 = s1 > 0.f ? s1 : -1.0e12f;
      const float a0 = __expf(s0 - mv.x) * rzv.x;   // exp err << f16 rounding
      const float a1 = __expf(s1 - mv.y) * rzv.y;
      const h2 hp = { (_Float16)a0, (_Float16)a1 };
      attLds[il * PAD + jp] = __builtin_bit_cast(unsigned int, hp);
    }
    // ---- stage hT tile [FD x TJ] (already f16-packed in global)
#pragma unroll
    for (int p = t; p < FD * TJ / 2; p += 256) {
      const int o  = p >> 5;
      const int jp = p & 31;
      hLds[o * PAD + jp] = hTu[(size_t)(b * FD + o) * (NN / 2) + (j0 >> 1) + jp];
    }
    __syncthreads();

#pragma unroll
    for (int c = 0; c < 2; ++c) {        // two K=32 chunks of this j-tile
      const int kc = c * 16;             // pair offset of chunk
      // ---- A fragment (16-bit 16x32 layout: v0..3 K={0..7|8..15}, v4..7 K+16)
      v8u au;
#pragma unroll
      for (int v = 0; v < 8; ++v) {
        const int kp = kc + (v & 3) + ((v >> 2) << 3) + hiHalf * 4;
        au[v] = attLds[(rg * 16 + lrow) * PAD + kp];
      }
      const v16h Af = __builtin_bit_cast(v16h, au);
      // ---- B fragments (16-bit 32x16 layout: v0..3 K={0..7|8..15}, v4..7 K+16)
      v8u bu0, bu1;
#pragma unroll
      for (int v = 0; v < 8; ++v) {
        const int kp = kc + (v & 3) + ((v >> 2) << 3) + hiHalf * 4;
        bu0[v] = hLds[((cg0)     * 16 + lrow) * PAD + kp];
        bu1[v] = hLds[((cg0 + 1) * 16 + lrow) * PAD + kp];
      }
      const v16h B0 = __builtin_bit_cast(v16h, bu0);
      const v16h B1 = __builtin_bit_cast(v16h, bu1);

      acc0 = __builtin_amdgcn_wmma_f32_16x16x32_f16(false, Af, false, B0, (short)0, acc0, false, false);
      acc1 = __builtin_amdgcn_wmma_f32_16x16x32_f16(false, Af, false, B1, (short)0, acc1, false, false);
    }
  }

  // ---- writeback with relu (C/D layout: VGPR r -> M = r + hiHalf*8, N = lane&15)
#pragma unroll
  for (int r = 0; r < 8; ++r) {
    const int row = i0 + rg * 16 + r + hiHalf * 8;
    const float v0c = acc0[r] > 0.f ? acc0[r] : 0.f;
    const float v1c = acc1[r] > 0.f ? acc1[r] : 0.f;
    out[((size_t)(b * NN + row)) * FD + cg0 * 16 + lrow]        = v0c;
    out[((size_t)(b * NN + row)) * FD + (cg0 + 1) * 16 + lrow]  = v1c;
  }
}

// ---------------------------------------------------------------------------
extern "C" void kernel_launch(void* const* d_in, const int* in_sizes, int n_in,
                              void* d_out, int out_size, void* d_ws, size_t ws_size,
                              hipStream_t stream) {
  const float* inp = (const float*)d_in[0];   // [B,N,64]
  const float* adj = (const float*)d_in[1];   // [B,N,N]
  const float* W   = (const float*)d_in[2];   // [64,64]
  const float* a   = (const float*)d_in[3];   // [128,1]
  float* out = (float*)d_out;                 // [B,N,64]

  char* ws = (char*)d_ws;
  _Float16* hT = (_Float16*)ws;                                   // 2 MB
  float* fsrc  = (float*)(ws + 2u * 1024 * 1024);                 // 64 KB
  float* fdst  = (float*)(ws + 2u * 1024 * 1024 +  64 * 1024);    // 64 KB
  float* cmax  = (float*)(ws + 2u * 1024 * 1024 + 128 * 1024);    // 64 KB
  float* crz   = (float*)(ws + 2u * 1024 * 1024 + 192 * 1024);    // 64 KB

  gat_proj_kernel<<<BATCH * NN, 64, 0, stream>>>(inp, W, a, hT, fsrc, fdst);
  gat_colstats_kernel<<<BATCH * NN / 256, 256, 0, stream>>>(adj, fsrc, fdst, cmax, crz);
  dim3 g3(NN / TI, BATCH);
  gat_attn_gemm_kernel<<<g3, 256, 0, stream>>>(adj, (const unsigned int*)hT,
                                               fsrc, fdst, cmax, crz, out);
}
